// kNNFit_30760555774711
// MI455X (gfx1250) — matrix operand order repaired and checked by reading, not compiled, not run on hardware
//
#include <hip/hip_runtime.h>
#include <hip/hip_bf16.h>
#include <stdint.h>

// MI455X (gfx1250) wave32 WMMA implementation.
// cross = (X2*W)^T @ X1 via V_WMMA_F32_16X16X32_F16.
// Workgroup = 8 waves (2x4 grid of 64x64 wave tiles => 128x256 WG tile).
// Per k-step the WG async-copies A(8KB)+B(16KB) fragments into double-
// buffered LDS with GLOBAL_LOAD_ASYNC_TO_LDS_B128 (ASYNCcnt), cutting L2
// traffic 2.7x vs per-wave global loads; WMMAs read via ds_load_b128.
// Pointer-swapped buffers + unroll(disable) + sched_barrier keep the loop
// body iteration-invariant so accumulators stay pinned (<256 VGPRs).

typedef __attribute__((ext_vector_type(16))) _Float16 v16h;
typedef __attribute__((ext_vector_type(8)))  float    v8f;

#define D_DIM  1024
#define N1_DIM 4096
#define N2_DIM 8192
#define KSTEPS (D_DIM / 32)     // 32 k-steps of K=32

#define ACHUNKS 8               // 16-row tiles per WG along n2 (128 rows)
#define BCHUNKS 16              // 16-col tiles per WG along n1 (256 cols)
#define CHUNKS  (ACHUNKS + BCHUNKS)
#define CHUNK_ELEMS 512         // one fragment: 32 lanes * 16 f16
#define STAGE_ELEMS (CHUNKS * CHUNK_ELEMS)   // 24 * 512 f16 = 24KB

// ---------------- small prep kernels ----------------

__global__ void norm_kernel(const float* __restrict__ W, float* __restrict__ nrm) {
  __shared__ float red[256];
  float a = 0.f;
  for (int d = threadIdx.x; d < D_DIM; d += 256) a += fabsf(W[d]);
  red[threadIdx.x] = a;
  __syncthreads();
  for (int s = 128; s > 0; s >>= 1) {
    if ((int)threadIdx.x < s) red[threadIdx.x] += red[threadIdx.x + s];
    __syncthreads();
  }
  if (threadIdx.x == 0) nrm[0] = red[0] + 1e-6f;
}

// sV[n] = sum_d W[d] * bits[d*N + n]
template <int NCOLS>
__global__ void wsum_kernel(const int* __restrict__ bits, const float* __restrict__ W,
                            float* __restrict__ sv) {
  int n = blockIdx.x * blockDim.x + threadIdx.x;
  if (n >= NCOLS) return;
  float acc = 0.f;
  for (int d = 0; d < D_DIM; ++d)
    acc += W[d] * (float)bits[d * NCOLS + n];
  sv[n] = acc;
}

// ---------------- fragment packing ----------------
// One thread per (tile, kstep, lane): each gather iteration reads a full
// contiguous 64B cacheline of bits per half-wave; one coalesced 32B store.
//
// A fragment (16x32 f16, M x K), ISA 7.12.2: lane holds row M=lane&15;
// lanes 0-15 -> K {0..7,16..23}, lanes 16-31 -> K {8..15,24..31}.
__global__ void pack_a_kernel(const int* __restrict__ x2bits, const float* __restrict__ W,
                              _Float16* __restrict__ Apack) {
  int t = blockIdx.x * blockDim.x + threadIdx.x;   // (tileM, ks, lane)
  int lane  = t & 31;
  int ks    = (t >> 5) & 31;
  int tileM = t >> 10;
  int m  = lane & 15;
  int kb = (lane < 16) ? 0 : 8;
  int n2 = tileM * 16 + m;
  v16h f;
#pragma unroll
  for (int i = 0; i < 16; ++i) {
    int k = (i < 8) ? (kb + i) : (kb + 8 + i);     // i>=8: kb+16+(i-8)
    int d = ks * 32 + k;
    f[i] = (_Float16)(W[d] * (float)x2bits[(size_t)d * N2_DIM + n2]);
  }
  *(v16h*)(Apack + (size_t)t * 16) = f;
}

// B fragment (32x16 f16, K x N): lane holds column N=lane&15;
// lanes 0-15 -> K {0..7,16..23}, lanes 16-31 -> K {8..15,24..31}.
__global__ void pack_b_kernel(const int* __restrict__ x1bits,
                              _Float16* __restrict__ Bpack) {
  int t = blockIdx.x * blockDim.x + threadIdx.x;   // (tileN, ks, lane)
  int lane  = t & 31;
  int ks    = (t >> 5) & 31;
  int tileN = t >> 10;
  int n  = lane & 15;
  int kb = (lane < 16) ? 0 : 8;
  int n1 = tileN * 16 + n;
  v16h f;
#pragma unroll
  for (int i = 0; i < 16; ++i) {
    int d = ks * 32 + ((i < 8) ? (kb + i) : (kb + 8 + i));
    f[i] = (_Float16)(float)x1bits[(size_t)d * N1_DIM + n1];
  }
  *(v16h*)(Bpack + (size_t)t * 16) = f;
}

// ---------------- fused GEMM + epilogue ----------------

__device__ __forceinline__ float Ffun(float x, float jc, float kc) {
  x = fminf(fmaxf(x, 0.f), 1.f);
  float inner = x * (1.f - kc) / (kc * (1.f - 2.f * fabsf(x)) + 1.f);
  float t = inner * 2.f - 1.f;
  return (1.f + t * (1.f + jc) / (-jc * (1.f - 2.f * fabsf(t)) + 1.f)) * 0.5f;
}

__device__ __forceinline__ v8f wmma_f16(v16h a, v16h b, v8f c) {
  return __builtin_amdgcn_wmma_f32_16x16x32_f16(
      /*neg_a=*/false, a, /*neg_b=*/false, b,
      /*c_mod=*/(short)0, c, /*reuse_a=*/false, /*reuse_b=*/false);
}

__global__ void __launch_bounds__(256)
gemm_epilogue_kernel(const _Float16* __restrict__ Apack,
                     const _Float16* __restrict__ Bpack,
                     const float* __restrict__ s1,
                     const float* __restrict__ s2,
                     const float* __restrict__ Y,
                     const float* __restrict__ nrm,
                     const float* __restrict__ jp,
                     const float* __restrict__ kp,
                     float* __restrict__ partial) {
  __shared__ __align__(32) _Float16 smem[2][STAGE_ELEMS];   // 48KB double buffer

  const int tid  = threadIdx.x;
  const int lane = tid & 31;
  const int wave = tid >> 5;
  const int wm   = wave >> 2;                      // 0..1  (M position in WG)
  const int wn   = wave & 3;                       // 0..3  (N position in WG)

  const int bn1 = blockIdx.x & 15;                 // N1/256 = 16
  const int bn2 = blockIdx.x >> 4;                 // N2/128 = 64
  const int tM0 = bn2 * ACHUNKS;                   // first 16-row tile (n2)
  const int tN0 = bn1 * BCHUNKS;                   // first 16-col tile (n1)

  const int subw = tid & 63;                       // 16B unit within a chunk

  // Async-copy stage `ks` fragments into LDS at `stagebase`. 1536 16B units
  // over 256 threads: 6 GLOBAL_LOAD_ASYNC_TO_LDS_B128 per wave per stage.
  auto prefetch = [&](int ks, const _Float16* stagebase) {
#pragma unroll
    for (int r = 0; r < 6; ++r) {
      const int chunk = r * 4 + (tid >> 6);        // uniform within a wave
      uint32_t lds =
          (uint32_t)(uintptr_t)(stagebase + chunk * CHUNK_ELEMS + subw * 8);
      const char* gp;
      if (chunk < ACHUNKS)
        gp = (const char*)Apack +
             ((((size_t)(tM0 + chunk) * KSTEPS) + ks) << 10) + (subw << 4);
      else
        gp = (const char*)Bpack +
             ((((size_t)(tN0 + chunk - ACHUNKS) * KSTEPS) + ks) << 10) + (subw << 4);
      asm volatile("global_load_async_to_lds_b128 %0, %1, off"
                   :: "v"(lds), "v"(gp) : "memory");
    }
  };

  v8f c[4][4] = {};

  const _Float16* bufCur = &smem[0][0];
  const _Float16* bufNxt = &smem[1][0];

  prefetch(0, bufCur);

#pragma clang loop unroll(disable)
  for (int ks = 0; ks < KSTEPS; ++ks) {
    if (ks + 1 < KSTEPS) {
      prefetch(ks + 1, bufNxt);
      asm volatile("s_wait_asynccnt 6" ::: "memory");   // current stage landed
    } else {
      asm volatile("s_wait_asynccnt 0" ::: "memory");
    }
    __syncthreads();                                    // all waves' copies visible

    const _Float16* As = bufCur + (wm * 4) * CHUNK_ELEMS + lane * 16;
    const _Float16* Bs = bufCur + (ACHUNKS + wn * 4) * CHUNK_ELEMS + lane * 16;

    v16h a0 = *(const v16h*)(As);
    v16h a1 = *(const v16h*)(As + CHUNK_ELEMS);
    v16h a2 = *(const v16h*)(As + 2 * CHUNK_ELEMS);
    v16h a3 = *(const v16h*)(As + 3 * CHUNK_ELEMS);

    v16h b0 = *(const v16h*)(Bs);
    v16h b1 = *(const v16h*)(Bs + CHUNK_ELEMS);
    v16h b2 = *(const v16h*)(Bs + 2 * CHUNK_ELEMS);
    v16h b3 = *(const v16h*)(Bs + 3 * CHUNK_ELEMS);

    c[0][0] = wmma_f16(a0, b0, c[0][0]);
    c[0][1] = wmma_f16(a0, b1, c[0][1]);
    c[0][2] = wmma_f16(a0, b2, c[0][2]);
    c[0][3] = wmma_f16(a0, b3, c[0][3]);
    c[1][0] = wmma_f16(a1, b0, c[1][0]);
    c[1][1] = wmma_f16(a1, b1, c[1][1]);
    c[1][2] = wmma_f16(a1, b2, c[1][2]);
    c[1][3] = wmma_f16(a1, b3, c[1][3]);
    c[2][0] = wmma_f16(a2, b0, c[2][0]);
    c[2][1] = wmma_f16(a2, b1, c[2][1]);
    c[2][2] = wmma_f16(a2, b2, c[2][2]);
    c[2][3] = wmma_f16(a2, b3, c[2][3]);
    c[3][0] = wmma_f16(a3, b0, c[3][0]);
    c[3][1] = wmma_f16(a3, b1, c[3][1]);
    c[3][2] = wmma_f16(a3, b2, c[3][2]);
    c[3][3] = wmma_f16(a3, b3, c[3][3]);

    __syncthreads();               // done reading before next-next overwrite

    // Swap double buffers; fence scheduling so no cross-iteration pipelining
    // inflates register pressure past the 256-VGPR window.
    const _Float16* tswap = bufCur; bufCur = bufNxt; bufNxt = tswap;
    __builtin_amdgcn_sched_barrier(0);
  }

  // Fused epilogue: WHD -> F -> *Y[n2] -> reduce over n2 -> atomic partial.
  const float jc   = fminf(fmaxf(jp[0], -1.f), 1.f);
  const float kc   = fminf(fmaxf(kp[0], -1.f), 1.f);
  const float invn = 1.0f / nrm[0];
  const int col    = lane & 15;
  const int mhalf  = (lane >> 4) << 3;             // 0 or 8 (C layout halves)

#pragma unroll
  for (int nt = 0; nt < 4; ++nt) {
    const int n1 = (tN0 + wn * 4 + nt) * 16 + col;
    const float s1v = s1[n1];
    float acc = 0.f;
#pragma unroll
    for (int mt = 0; mt < 4; ++mt) {
      const int n2base = (tM0 + wm * 4 + mt) * 16 + mhalf;
#pragma unroll
      for (int r = 0; r < 8; ++r) {
        const int n2 = n2base + r;
        float whd = s2[n2] + s1v - 2.0f * c[mt][nt][r];
        float x = 1.0f - whd * invn;
        acc += Ffun(x, jc, kc) * Y[n2];
      }
    }
    acc += __shfl_xor(acc, 16, 32);                // add other M-half (lanes ^16)
    if (lane < 16) atomicAdd(&partial[n1], acc);
  }
}

__global__ void finalize_kernel(const float* __restrict__ partial,
                                const float* __restrict__ alphap,
                                const float* __restrict__ betap,
                                float* __restrict__ out) {
  int n = blockIdx.x * blockDim.x + threadIdx.x;
  if (n < N1_DIM)
    out[n] = betap[0] * (partial[n] * (1.0f / (float)N2_DIM) + alphap[0]);
}

// ---------------- launcher ----------------

extern "C" void kernel_launch(void* const* d_in, const int* in_sizes, int n_in,
                              void* d_out, int out_size, void* d_ws, size_t ws_size,
                              hipStream_t stream) {
  const int*   x1     = (const int*)d_in[0];   // [D, N1] bits
  const int*   x2     = (const int*)d_in[1];   // [D, N2] bits
  /* d_in[2] (byDimBlock) unused */
  const float* Y      = (const float*)d_in[3]; // [N2]
  const float* W      = (const float*)d_in[4]; // [D]
  const float* alphap = (const float*)d_in[5];
  const float* betap  = (const float*)d_in[6];
  const float* kp     = (const float*)d_in[7];
  const float* jp     = (const float*)d_in[8];
  float* out = (float*)d_out;

  char* ws = (char*)d_ws;
  float*    partial = (float*)(ws);                 // N1 floats
  float*    s1      = (float*)(ws + 16384);         // N1 floats
  float*    s2      = (float*)(ws + 32768);         // N2 floats
  float*    nrm     = (float*)(ws + 65536);         // 1 float
  _Float16* Apack   = (_Float16*)(ws + 131072);                                   // N2*D f16 (16 MB)
  _Float16* Bpack   = (_Float16*)(ws + 131072 + (size_t)N2_DIM * D_DIM * 2);      // N1*D f16 (8 MB)

  hipMemsetAsync(partial, 0, N1_DIM * sizeof(float), stream);

  norm_kernel<<<1, 256, 0, stream>>>(W, nrm);
  wsum_kernel<N1_DIM><<<N1_DIM / 256, 256, 0, stream>>>(x1, W, s1);
  wsum_kernel<N2_DIM><<<N2_DIM / 256, 256, 0, stream>>>(x2, W, s2);

  // (tiles * ksteps * 32 lanes) threads
  pack_a_kernel<<<(N2_DIM / 16) * KSTEPS * 32 / 256, 256, 0, stream>>>(x2, W, Apack);
  pack_b_kernel<<<(N1_DIM / 16) * KSTEPS * 32 / 256, 256, 0, stream>>>(x1, Bpack);

  // WG tile 128(n2) x 256(n1): (N2/128)*(N1/256) = 1024 workgroups.
  gemm_epilogue_kernel<<<(N2_DIM / 128) * (N1_DIM / 256), 256, 0, stream>>>(
      Apack, Bpack, s1, s2, Y, nrm, jp, kp, partial);

  finalize_kernel<<<N1_DIM / 256, 256, 0, stream>>>(partial, alphap, betap, out);
}
